// Encoder_39187281609247
// MI455X (gfx1250) — matrix operand order repaired and checked by reading, not compiled, not run on hardware
//
#include <hip/hip_runtime.h>
#include <math.h>

// ---------------------------------------------------------------------------
// Transformer encoder layer for MI455X (gfx1250), wave32, WMMA bf16 path.
// B=2, T=2048, D=1024, H=16, DH=64, DFF=4096.
// ---------------------------------------------------------------------------

#define B_  2
#define T_  2048
#define D_  1024
#define H_  16
#define DH_ 64
#define DFF_ 4096
#define BT_ (B_ * T_)

typedef __attribute__((ext_vector_type(16))) __bf16 v16bf;
typedef __attribute__((ext_vector_type(8)))  float  v8f;

struct Frag32B { uint4 lo; uint4 hi; };
static_assert(sizeof(Frag32B) == 32, "frag size");
static_assert(sizeof(v16bf) == 32, "v16bf size");

static __device__ __forceinline__ v16bf make_frag(uint4 lo, uint4 hi) {
    Frag32B t{lo, hi};
    return __builtin_bit_cast(v16bf, t);
}

static __device__ __forceinline__ unsigned short f2bf(float f) {
    unsigned u = __builtin_bit_cast(unsigned, f);
    u += 0x7FFFu + ((u >> 16) & 1u);   // round-to-nearest-even
    return (unsigned short)(u >> 16);
}

static __device__ __forceinline__ v8f vzero() {
    v8f z;
    #pragma unroll
    for (int i = 0; i < 8; ++i) z[i] = 0.0f;
    return z;
}

static __device__ __forceinline__ v8f wmma_bf16(v16bf a, v16bf b, v8f c) {
    // (neg_a, A, neg_b, B, c_mod, C, reuse_a, reuse_b)
    return __builtin_amdgcn_wmma_f32_16x16x32_bf16(false, a, false, b,
                                                   (short)0, c, false, false);
}

// --- DPP16 cross-lane helpers (no LDS round trip, unlike ds_bpermute) -------
// ctrl: 0xB1 quad_perm(1,0,3,2)=xor1, 0x4E quad_perm(2,3,0,1)=xor2,
//       0x141 row_half_mirror (4-distance once quads uniform),
//       0x140 row_mirror (8-distance within 16-lane row).
template <int CTRL>
static __device__ __forceinline__ float dpp_f(float x) {
    int r = __builtin_amdgcn_update_dpp(0, __builtin_bit_cast(int, x),
                                        CTRL, 0xF, 0xF, true);
    return __builtin_bit_cast(float, r);
}

static __device__ __forceinline__ float rmax16(float x) {
    x = fmaxf(x, dpp_f<0xB1>(x));
    x = fmaxf(x, dpp_f<0x4E>(x));
    x = fmaxf(x, dpp_f<0x141>(x));
    x = fmaxf(x, dpp_f<0x140>(x));
    return x;
}

static __device__ __forceinline__ float rsum16(float x) {
    x += dpp_f<0xB1>(x);
    x += dpp_f<0x4E>(x);
    x += dpp_f<0x141>(x);
    x += dpp_f<0x140>(x);
    return x;
}

// ---------------------------------------------------------------------------
// fp32 -> bf16 conversion (weights)
// ---------------------------------------------------------------------------
__global__ void cvt_f32_bf16(const float* __restrict__ in,
                             unsigned short* __restrict__ out, int n) {
    int i = blockIdx.x * blockDim.x + threadIdx.x;
    if (i < n) out[i] = f2bf(in[i]);
}

// ---------------------------------------------------------------------------
// LayerNorm over rows of [BT, D]; output bf16.
// ---------------------------------------------------------------------------
__global__ __launch_bounds__(256)
void ln_kernel(const float* __restrict__ x, const float* __restrict__ g,
               const float* __restrict__ b, unsigned short* __restrict__ out) {
    __shared__ float redS[8], redS2[8];
    const int row = blockIdx.x, tid = threadIdx.x;
    const float* xr = x + (size_t)row * D_;
    float f[4], s = 0.f, s2 = 0.f;
    #pragma unroll
    for (int i = 0; i < 4; ++i) {
        f[i] = xr[tid + 256 * i];
        s += f[i];
        s2 += f[i] * f[i];
    }
    #pragma unroll
    for (int d = 1; d < 32; d <<= 1) {
        s  += __shfl_xor(s,  d);
        s2 += __shfl_xor(s2, d);
    }
    if ((tid & 31) == 0) { redS[tid >> 5] = s; redS2[tid >> 5] = s2; }
    __syncthreads();
    s = 0.f; s2 = 0.f;
    #pragma unroll
    for (int i = 0; i < 8; ++i) { s += redS[i]; s2 += redS2[i]; }
    const float mean = s * (1.0f / D_);
    const float var  = s2 * (1.0f / D_) - mean * mean;
    const float rstd = rsqrtf(var + 1e-5f);
    #pragma unroll
    for (int i = 0; i < 4; ++i) {
        int c = tid + 256 * i;
        out[(size_t)row * D_ + c] = f2bf((f[i] - mean) * rstd * g[c] + b[c]);
    }
}

// ---------------------------------------------------------------------------
// Tiled bf16 WMMA GEMM:  out = act(A[MxK] @ B[KxN] + bias) (+ residual)
// MODE 0: bf16 out;  MODE 1: exact GELU, bf16 out;  MODE 2: +residual, f32 out
// Block: 256 threads (8 waves). Tile: 128x64, K-step 32, double-buffered LDS,
// one barrier per K-step, global_prefetch one stage ahead.
// ---------------------------------------------------------------------------
template <int MODE>
__global__ __launch_bounds__(256)
void gemm_bf16(const unsigned short* __restrict__ A,
               const unsigned short* __restrict__ Bw,
               const float* __restrict__ bias,
               const float* __restrict__ res,
               void* __restrict__ out, int M, int N, int K) {
    __shared__ __align__(16) unsigned short As[2][128 * 40];  // [row][k] padded
    __shared__ __align__(16) unsigned short Bs[2][64 * 40];   // [col][k] (B^T)
    const int tid = threadIdx.x;
    const int lane = tid & 31, wave = tid >> 5;
    const int m0 = blockIdx.y * 128, n0 = blockIdx.x * 64;

    const unsigned* Ag = (const unsigned*)A;
    const unsigned* Bg = (const unsigned*)Bw;

    auto load_tiles = [&](int buf, int k0) {
        // A tile 128x32 halfs = 2048 dwords, row-major into LDS
        #pragma unroll
        for (int i = 0; i < 8; ++i) {
            int idx = tid + i * 256;
            int r = idx >> 4, cp = idx & 15;
            unsigned d = Ag[(size_t)(m0 + r) * (K >> 1) + (k0 >> 1) + cp];
            *(unsigned*)&As[buf][r * 40 + cp * 2] = d;
        }
        // B tile 32x64 halfs -> transposed into LDS (k-contiguous per column)
        #pragma unroll
        for (int i = 0; i < 4; ++i) {
            int idx = tid + i * 256;
            int r = idx >> 5, cp = idx & 31;
            unsigned d = Bg[(size_t)(k0 + r) * (N >> 1) + (n0 >> 1) + cp];
            Bs[buf][(cp * 2    ) * 40 + r] = (unsigned short)(d & 0xFFFFu);
            Bs[buf][(cp * 2 + 1) * 40 + r] = (unsigned short)(d >> 16);
        }
        // prefetch one K-step further ahead (global_prefetch_b8 path)
        if (k0 + 32 < K) {
            int ra = tid >> 1;  // 128 rows, 2 threads/row (each line covers 32B)
            __builtin_prefetch(&Ag[(size_t)(m0 + ra) * (K >> 1) +
                                   ((k0 + 32) >> 1) + (tid & 1) * 8], 0, 0);
            if (tid < 32) {
                __builtin_prefetch(&Bg[(size_t)(k0 + 32 + tid) * (N >> 1) +
                                       (n0 >> 1)], 0, 0);
            }
        }
    };

    v8f acc[4];
    #pragma unroll
    for (int i = 0; i < 4; ++i) acc[i] = vzero();

    load_tiles(0, 0);
    int buf = 0;
    for (int k0 = 0; k0 < K; k0 += 32) {
        __syncthreads();                       // tiles in `buf` are visible
        if (k0 + 32 < K) load_tiles(buf ^ 1, k0 + 32);

        // A fragment: lane m = lane&15, contiguous K order (k = 16*(lane>>4) + s)
        const int mrow = wave * 16 + (lane & 15);
        const int kb = (lane >> 4) * 16;
        v16bf afrag = make_frag(*(const uint4*)&As[buf][mrow * 40 + kb],
                                *(const uint4*)&As[buf][mrow * 40 + kb + 8]);
        const int kg = (lane >> 4) * 16;
        #pragma unroll
        for (int ns = 0; ns < 4; ++ns) {
            int ncol = ns * 16 + (lane & 15);
            const uint4* Bp = (const uint4*)&Bs[buf][ncol * 40 + kg];
            v16bf bfrag = make_frag(Bp[0], Bp[1]);
            acc[ns] = wmma_bf16(afrag, bfrag, acc[ns]);
        }
        buf ^= 1;
    }

    // Epilogue. C layout: VGPR v -> row v + 8*(lane>=16), col = lane&15.
    const int hi = lane >> 4, nl = lane & 15;
    #pragma unroll
    for (int ns = 0; ns < 4; ++ns) {
        int col = n0 + ns * 16 + nl;
        float bval = bias[col];
        #pragma unroll
        for (int v = 0; v < 8; ++v) {
            int row = m0 + wave * 16 + v + 8 * hi;
            float x = acc[ns][v] + bval;
            if constexpr (MODE == 0) {
                ((unsigned short*)out)[(size_t)row * N + col] = f2bf(x);
            } else if constexpr (MODE == 1) {
                float gl = 0.5f * x * (1.0f + erff(x * 0.70710678118654752f));
                ((unsigned short*)out)[(size_t)row * N + col] = f2bf(gl);
            } else {
                ((float*)out)[(size_t)row * N + col] =
                    x + res[(size_t)row * N + col];
            }
        }
    }
}

// ---------------------------------------------------------------------------
// Flash attention. Q,K,V bf16 [BT, D] (head h occupies cols h*64..h*64+63).
// Grid: (T/64, H, B). Block: 128 threads (4 waves), each wave owns 16 q rows.
// ---------------------------------------------------------------------------
__global__ __launch_bounds__(128)
void flash_kernel(const unsigned short* __restrict__ Q,
                  const unsigned short* __restrict__ Km,
                  const unsigned short* __restrict__ Vm,
                  unsigned short* __restrict__ O) {
    __shared__ __align__(16) unsigned short Vs[64 * 72];       // V^T: [dh][key]
    __shared__ __align__(16) unsigned short Ps[4][16 * 72];    // per-wave P

    const int tid = threadIdx.x, lane = tid & 31, wave = tid >> 5;
    const int h = blockIdx.y;
    const size_t base = ((size_t)blockIdx.z * T_) * D_ + (size_t)h * DH_;
    const int q0 = blockIdx.x * 64 + wave * 16;
    const float scale = 0.125f;  // 1/sqrt(64)

    const int mrow = lane & 15;
    const int kb = (lane >> 4) * 16;
    const int hi = lane >> 4, nl = lane & 15;

    // Q fragments (contraction over d_h, two 32-wide chunks), direct from global
    v16bf qf[2];
    #pragma unroll
    for (int kc = 0; kc < 2; ++kc) {
        const unsigned short* p = Q + base + (size_t)(q0 + mrow) * D_ + kc * 32;
        qf[kc] = make_frag(*(const uint4*)(p + kb), *(const uint4*)(p + kb + 8));
    }

    v8f o[4];
    #pragma unroll
    for (int t = 0; t < 4; ++t) o[t] = vzero();
    float mr[8], lr[8];
    #pragma unroll
    for (int v = 0; v < 8; ++v) { mr[v] = -1e30f; lr[v] = 0.0f; }

    for (int j0 = 0; j0 < T_; j0 += 64) {
        // stage V tile transposed into LDS (cooperative)
        #pragma unroll
        for (int i = 0; i < 16; ++i) {
            int idx = tid + i * 128;           // 2048 dwords
            int key = idx >> 5, dp = idx & 31;
            unsigned d = *(const unsigned*)(Vm + base + (size_t)(j0 + key) * D_ + dp * 2);
            Vs[(dp * 2    ) * 72 + key] = (unsigned short)(d & 0xFFFFu);
            Vs[(dp * 2 + 1) * 72 + key] = (unsigned short)(d >> 16);
            if (i < 2 && j0 + 64 < T_) {
                __builtin_prefetch(
                    Vm + base + (size_t)(j0 + 64 + key) * D_ + dp * 2, 0, 0);
            }
        }
        __syncthreads();

        // S = Q @ K^T : contraction over d_h (memory-contiguous -> direct B frags)
        v8f s[4];
        #pragma unroll
        for (int ns = 0; ns < 4; ++ns) {
            v8f sacc = vzero();
            int key = j0 + ns * 16 + nl;
            #pragma unroll
            for (int kc = 0; kc < 2; ++kc) {
                const unsigned short* p =
                    Km + base + (size_t)key * D_ + kc * 32 + hi * 16;
                v16bf bf = make_frag(*(const uint4*)p, *(const uint4*)(p + 8));
                sacc = wmma_bf16(qf[kc], bf, sacc);
            }
            s[ns] = sacc;
        }

        // online softmax (fp32). Row of C layout lives in a 16-lane group;
        // reductions use DPP16 (quad_perm / row mirrors), no LDS traffic.
        #pragma unroll
        for (int ns = 0; ns < 4; ++ns)
            #pragma unroll
            for (int v = 0; v < 8; ++v) s[ns][v] *= scale;

        #pragma unroll
        for (int v = 0; v < 8; ++v) {
            float mv = fmaxf(fmaxf(s[0][v], s[1][v]), fmaxf(s[2][v], s[3][v]));
            mv = rmax16(mv);
            float mnew = fmaxf(mr[v], mv);
            float corr = __expf(mr[v] - mnew);
            mr[v] = mnew;
            float rs = 0.0f;
            #pragma unroll
            for (int ns = 0; ns < 4; ++ns) {
                float p = __expf(s[ns][v] - mnew);
                s[ns][v] = p;
                rs += p;
            }
            rs = rsum16(rs);
            lr[v] = lr[v] * corr + rs;
            #pragma unroll
            for (int t = 0; t < 4; ++t) o[t][v] *= corr;
        }

        // P (C layout) -> per-wave LDS -> A layout
        #pragma unroll
        for (int ns = 0; ns < 4; ++ns)
            #pragma unroll
            for (int v = 0; v < 8; ++v)
                Ps[wave][(v + 8 * hi) * 72 + ns * 16 + nl] = f2bf(s[ns][v]);

        v16bf pf[2];
        #pragma unroll
        for (int kc = 0; kc < 2; ++kc) {
            const unsigned short* p = &Ps[wave][mrow * 72 + kc * 32 + kb];
            pf[kc] = make_frag(*(const uint4*)p, *(const uint4*)(p + 8));
        }
        // O += P @ V : contraction over keys via transposed Vs
        #pragma unroll
        for (int t = 0; t < 4; ++t) {
            #pragma unroll
            for (int kc = 0; kc < 2; ++kc) {
                const unsigned short* p = &Vs[(t * 16 + nl) * 72 + kc * 32 + hi * 16];
                v16bf vf = make_frag(*(const uint4*)p, *(const uint4*)(p + 8));
                o[t] = wmma_bf16(pf[kc], vf, o[t]);
            }
        }
        __syncthreads();  // protect Vs before next tile
    }

    // normalize and write bf16
    #pragma unroll
    for (int v = 0; v < 8; ++v) {
        float inv = 1.0f / lr[v];
        int row = q0 + v + 8 * hi;
        #pragma unroll
        for (int t = 0; t < 4; ++t) {
            int col = t * 16 + nl;
            O[base + (size_t)row * D_ + col] = f2bf(o[t][v] * inv);
        }
    }
}

// ---------------------------------------------------------------------------
// Host-side orchestration
// ---------------------------------------------------------------------------
extern "C" void kernel_launch(void* const* d_in, const int* in_sizes, int n_in,
                              void* d_out, int out_size, void* d_ws, size_t ws_size,
                              hipStream_t stream) {
    const float* x    = (const float*)d_in[0];
    const float* Wq   = (const float*)d_in[1];
    const float* bq   = (const float*)d_in[2];
    const float* Wk   = (const float*)d_in[3];
    const float* bk   = (const float*)d_in[4];
    const float* Wv   = (const float*)d_in[5];
    const float* bv   = (const float*)d_in[6];
    const float* Wo   = (const float*)d_in[7];
    const float* bo   = (const float*)d_in[8];
    const float* W1   = (const float*)d_in[9];
    const float* b1   = (const float*)d_in[10];
    const float* W2   = (const float*)d_in[11];
    const float* b2   = (const float*)d_in[12];
    const float* ln1g = (const float*)d_in[13];
    const float* ln1b = (const float*)d_in[14];
    const float* ln2g = (const float*)d_in[15];
    const float* ln2b = (const float*)d_in[16];

    char* ws = (char*)d_ws;
    size_t off = 0;
    auto carve = [&](size_t bytes) {
        void* p = ws + off;
        off += (bytes + 255) & ~(size_t)255;
        return p;
    };
    unsigned short* wqB  = (unsigned short*)carve((size_t)D_ * D_ * 2);
    unsigned short* wkB  = (unsigned short*)carve((size_t)D_ * D_ * 2);
    unsigned short* wvB  = (unsigned short*)carve((size_t)D_ * D_ * 2);
    unsigned short* woB  = (unsigned short*)carve((size_t)D_ * D_ * 2);
    unsigned short* w1B  = (unsigned short*)carve((size_t)D_ * DFF_ * 2);
    unsigned short* w2B  = (unsigned short*)carve((size_t)DFF_ * D_ * 2);
    unsigned short* hB   = (unsigned short*)carve((size_t)BT_ * D_ * 2);
    unsigned short* qB   = (unsigned short*)carve((size_t)BT_ * D_ * 2);
    unsigned short* kB   = (unsigned short*)carve((size_t)BT_ * D_ * 2);
    unsigned short* vB   = (unsigned short*)carve((size_t)BT_ * D_ * 2);
    unsigned short* attB = (unsigned short*)carve((size_t)BT_ * D_ * 2);
    float*          x2   = (float*)carve((size_t)BT_ * D_ * 4);
    unsigned short* h2B  = (unsigned short*)carve((size_t)BT_ * D_ * 2);
    unsigned short* ff1B = (unsigned short*)carve((size_t)BT_ * DFF_ * 2);
    (void)ws_size; (void)in_sizes; (void)n_in; (void)out_size;

    // 1) weights -> bf16
    const int nDD = D_ * D_, nDF = D_ * DFF_;
    cvt_f32_bf16<<<(nDD + 255) / 256, 256, 0, stream>>>(Wq, wqB, nDD);
    cvt_f32_bf16<<<(nDD + 255) / 256, 256, 0, stream>>>(Wk, wkB, nDD);
    cvt_f32_bf16<<<(nDD + 255) / 256, 256, 0, stream>>>(Wv, wvB, nDD);
    cvt_f32_bf16<<<(nDD + 255) / 256, 256, 0, stream>>>(Wo, woB, nDD);
    cvt_f32_bf16<<<(nDF + 255) / 256, 256, 0, stream>>>(W1, w1B, nDF);
    cvt_f32_bf16<<<(nDF + 255) / 256, 256, 0, stream>>>(W2, w2B, nDF);

    // 2) LN1
    ln_kernel<<<BT_, 256, 0, stream>>>(x, ln1g, ln1b, hB);

    // 3) Q, K, V projections (bf16 out)
    dim3 gDD(D_ / 64, BT_ / 128);
    gemm_bf16<0><<<gDD, 256, 0, stream>>>(hB, wqB, bq, nullptr, qB, BT_, D_, D_);
    gemm_bf16<0><<<gDD, 256, 0, stream>>>(hB, wkB, bk, nullptr, kB, BT_, D_, D_);
    gemm_bf16<0><<<gDD, 256, 0, stream>>>(hB, wvB, bv, nullptr, vB, BT_, D_, D_);

    // 4) flash attention
    flash_kernel<<<dim3(T_ / 64, H_, B_), 128, 0, stream>>>(qB, kB, vB, attB);

    // 5) output projection + residual (fp32)
    gemm_bf16<2><<<gDD, 256, 0, stream>>>(attB, woB, bo, x, x2, BT_, D_, D_);

    // 6) LN2
    ln_kernel<<<BT_, 256, 0, stream>>>(x2, ln2g, ln2b, h2B);

    // 7) FFN up + exact GELU (bf16 out)
    dim3 gDF(DFF_ / 64, BT_ / 128);
    gemm_bf16<1><<<gDF, 256, 0, stream>>>(h2B, w1B, b1, nullptr, ff1B, BT_, DFF_, D_);

    // 8) FFN down + residual -> d_out (fp32)
    gemm_bf16<2><<<gDD, 256, 0, stream>>>(ff1B, w2B, b2, x2, (float*)d_out,
                                          BT_, D_, DFF_);
}
